// PrismGATLayer_82497731821636
// MI455X (gfx1250) — compile-verified, matrix operand order checked by reading, not run on hardware
//
#include <hip/hip_runtime.h>
#include <math.h>

// ---------- types ----------
typedef __attribute__((ext_vector_type(16))) __bf16        v16bf;
typedef __attribute__((ext_vector_type(8)))  float         v8f;
typedef __attribute__((ext_vector_type(4)))  unsigned int  u32x4;
typedef __attribute__((ext_vector_type(4)))  int           i32x4;
typedef __attribute__((ext_vector_type(4)))  float         f32x4;

struct U32x4Pair { u32x4 lo; u32x4 hi; };

// explicit address-space pointer types for clean global/ds codegen
typedef const __attribute__((address_space(1))) u32x4*          gvec_cp;
typedef const __attribute__((address_space(3))) u32x4*          lvec_cp;
typedef __attribute__((address_space(1))) i32x4*                g_i32x4_p;
typedef __attribute__((address_space(3))) i32x4*                l_i32x4_p;
typedef const __attribute__((address_space(1))) float*          gf_cp;
typedef const __attribute__((address_space(1))) f32x4*          gf4_cp;
typedef const __attribute__((address_space(1))) int*            gi_cp;
typedef const __attribute__((address_space(1))) unsigned short* gu16_cp;
typedef __attribute__((address_space(1))) unsigned short*       gu16_p;
typedef __attribute__((address_space(1))) float*                gf_p;

// native RNE f32 -> bf16 (v_cvt_pk_bf16_f32 on gfx1250)
__device__ __forceinline__ unsigned short f2bf(float f) {
  return __builtin_bit_cast(unsigned short, (__bf16)f);
}

// ---------- optional CDNA5 async global->LDS path ----------
#if defined(__gfx1250__) &&                                                    \
    __has_builtin(__builtin_amdgcn_global_load_async_to_lds_b128) &&           \
    __has_builtin(__builtin_amdgcn_s_wait_asynccnt)
#define HAS_ASYNC_LDS 1
#endif

__device__ __forceinline__ void async_copy_b128(void* lds_dst, const void* gsrc) {
#ifdef HAS_ASYNC_LDS
  __builtin_amdgcn_global_load_async_to_lds_b128(
      (g_i32x4_p)gsrc, (l_i32x4_p)lds_dst, 0, 0);
#else
  *(u32x4*)lds_dst = *(const u32x4*)gsrc;
#endif
}

__device__ __forceinline__ void async_wait() {
#ifdef HAS_ASYNC_LDS
  __builtin_amdgcn_s_wait_asynccnt(0);
#endif
}

// ---------- WMMA fragment loads ----------
// Lane l: row = row0 + (l&15), half = l>>4.
// elems 0..7  <- K = k0 + half*8 + j      (VGPRs 0..3)
// elems 8..15 <- K = k0 + 16 + half*8 + j (VGPRs 4..7)
__device__ __forceinline__ v16bf load_frag_l(const unsigned short* p, int stride,
                                             int row0, int k0) {
  int lane = threadIdx.x & 31;
  const unsigned short* base =
      p + (size_t)(row0 + (lane & 15)) * stride + k0 + ((lane >> 4) << 3);
  U32x4Pair pr;
  pr.lo = ((lvec_cp)(const void*)base)[0];
  pr.hi = ((lvec_cp)(const void*)(base + 16))[0];
  return __builtin_bit_cast(v16bf, pr);
}

__device__ __forceinline__ v16bf load_frag_g(const unsigned short* p, int stride,
                                             int row0, int k0) {
  int lane = threadIdx.x & 31;
  const unsigned short* base =
      p + (size_t)(row0 + (lane & 15)) * stride + k0 + ((lane >> 4) << 3);
  U32x4Pair pr;
  pr.lo = ((gvec_cp)(const void*)base)[0];
  pr.hi = ((gvec_cp)(const void*)(base + 16))[0];
  return __builtin_bit_cast(v16bf, pr);
}

#define WMMA_BF16(A, B, C)                                                     \
  __builtin_amdgcn_wmma_f32_16x16x32_bf16(false, (A), false, (B), (short)0,    \
                                          (C), false, false)

#define ZERO8 {0.f, 0.f, 0.f, 0.f, 0.f, 0.f, 0.f, 0.f}

// ---------- kernel 0: weight convert + transpose (f32 [K,N] -> bf16 [N,K]) ----------
__global__ __launch_bounds__(256) void wconv_kernel(const float* __restrict__ src,
                                                    unsigned short* __restrict__ dst,
                                                    int K, int N) {
  int i = blockIdx.x * 256 + threadIdx.x;
  if (i < K * N) {
    int k = i / N, n = i - k * N;
    ((gu16_p)(void*)dst)[n * K + k] = f2bf(((gf_cp)(const void*)src)[i]);
  }
}

// ---------- kernel 1: fused QKV projections ----------
// X: [nblocks*128, 128] f32 staged as bf16 in LDS; up to 4 GEMMs vs col-major
// bf16 weights (B frags streamed from L2). Output: [B, H, N, 16] bf16.
__global__ __launch_bounds__(256) void proj_qkv_kernel(
    const float* __restrict__ X,
    const unsigned short* w0, const float* b0, unsigned short* o0,
    const unsigned short* w1, const float* b1, unsigned short* o1,
    const unsigned short* w2, const float* b2, unsigned short* o2,
    const unsigned short* w3, const float* b3, unsigned short* o3, int nw) {
  __shared__ unsigned short As[128 * 136];  // 34.8 KB, stride 136 (16B aligned)
  const unsigned short* ws[4] = {w0, w1, w2, w3};
  const float* bs[4] = {b0, b1, b2, b3};
  unsigned short* os[4] = {o0, o1, o2, o3};

  const int t = threadIdx.x;
  const int row0 = blockIdx.x * 128;
  gf4_cp x4 = (gf4_cp)(const void*)(X + (size_t)row0 * 128);
  for (int i = t; i < 128 * 32; i += 256) {
    f32x4 v = x4[i];
    int r = i >> 5, c = (i & 31) << 2;
    unsigned short* p = &As[r * 136 + c];
    p[0] = f2bf(v.x); p[1] = f2bf(v.y); p[2] = f2bf(v.z); p[3] = f2bf(v.w);
  }
  __syncthreads();

  const int wave = t >> 5, lane = t & 31;
  const int tq = wave;  // 8 waves -> 8 row tiles
  v16bf af[4];
#pragma unroll
  for (int kb = 0; kb < 4; ++kb) af[kb] = load_frag_l(As, 136, tq * 16, kb * 32);

  for (int w = 0; w < nw; ++w) {
    const unsigned short* wt = ws[w];
    gf_cp bw = (gf_cp)(const void*)bs[w];
    gu16_p op = (gu16_p)(void*)os[w];
    for (int tc = 0; tc < 8; ++tc) {
      v8f acc = ZERO8;
#pragma unroll
      for (int kb = 0; kb < 4; ++kb) {
        v16bf bfrag = load_frag_g(wt, 128, tc * 16, kb * 32);
        acc = WMMA_BF16(af[kb], bfrag, acc);
      }
      int d = lane & 15;
      float bias = bw[tc * 16 + d];
#pragma unroll
      for (int r = 0; r < 8; ++r) {
        int row = row0 + tq * 16 + r + ((lane >> 4) << 3);
        int bb = row >> 7, n = row & 127;
        // [B,H,N,16], h == tc
        op[(size_t)bb * 16384 + tc * 2048 + n * 16 + d] = f2bf(acc[r] + bias);
      }
    }
  }
}

// ---------- kernel 2: attention ----------
// One block per (b, h, q-quarter of 32 rows). Q/K/V bf16 in [B,H,128,16].
// Writes merged-head output [B*128, 128] bf16.
__global__ __launch_bounds__(256) void attn_kernel(
    const unsigned short* __restrict__ Q, const unsigned short* __restrict__ Kg,
    const unsigned short* __restrict__ V, const int* __restrict__ adj,
    const int* __restrict__ mask, unsigned short* __restrict__ Out, int useAdj) {
  __shared__ unsigned short Qs[32 * 32];    // 2 KB (K zero-padded to 32)
  __shared__ unsigned short Ks[128 * 32];   // 8 KB
  __shared__ unsigned short Vt[16 * 136];   // 4.35 KB (transposed V)
  __shared__ float Lg[32 * 129];            // 16.5 KB f32 logits
  __shared__ unsigned short Ps[32 * 136];   // 8.7 KB bf16 probabilities

  const int t = threadIdx.x;
  const int blk = blockIdx.x;
  const int bh = blk >> 2, qh = blk & 3;
  const int b = bh >> 3, h = bh & 7;
  const size_t base = (size_t)bh * 2048;  // elements in [B,H,128,16]
  gi_cp maskg = (gi_cp)(const void*)mask;
  gi_cp adjg = (gi_cp)(const void*)adj;
  gu16_cp vg = (gu16_cp)(const void*)V;

  const u32x4 z4 = {0u, 0u, 0u, 0u};
  const u32x4* q4 = (const u32x4*)(Q + base + (size_t)qh * 32 * 16);
  for (int i = t; i < 64; i += 256) {
    int r = i >> 1, part = i & 1;
    async_copy_b128(&Qs[r * 32 + part * 8], &q4[i]);
    *(u32x4*)&Qs[r * 32 + 16 + part * 8] = z4;
  }
  const u32x4* k4 = (const u32x4*)(Kg + base);
  for (int i = t; i < 256; i += 256) {
    int r = i >> 1, part = i & 1;
    async_copy_b128(&Ks[r * 32 + part * 8], &k4[i]);
    *(u32x4*)&Ks[r * 32 + 16 + part * 8] = z4;
  }
  for (int i = t; i < 2048; i += 256) {
    int n = i >> 4, d = i & 15;
    Vt[d * 136 + n] = vg[base + i];
  }
  async_wait();
  __syncthreads();

  const int wave = t >> 5, lane = t & 31;
  // ---- logits = Q K^T * 0.25 + masks ----
  {
    const int tq = wave >> 2;        // 0..1
    const int tc0 = (wave & 3) * 2;  // 0,2,4,6
    v16bf aq = load_frag_l(Qs, 32, tq * 16, 0);
    for (int tc = tc0; tc < tc0 + 2; ++tc) {
      v16bf bk = load_frag_l(Ks, 32, tc * 16, 0);
      v8f acc = ZERO8;
      acc = WMMA_BF16(aq, bk, acc);
      int kidx = tc * 16 + (lane & 15);
      float mterm = (1.0f - (float)maskg[b * 128 + kidx]) * -1e9f;
#pragma unroll
      for (int r = 0; r < 8; ++r) {
        int ql = tq * 16 + r + ((lane >> 4) << 3);
        int qg = qh * 32 + ql;
        float lg = acc[r] * 0.25f + mterm;
        if (useAdj)
          lg += (1.0f - (float)adjg[((size_t)b * 128 + qg) * 128 + kidx]) * -1e9f;
        Lg[ql * 129 + kidx] = lg;
      }
    }
  }
  __syncthreads();
  // ---- softmax (one thread per row) ----
  if (t < 32) {
    float mx = -3.0e38f;
    for (int k = 0; k < 128; ++k) mx = fmaxf(mx, Lg[t * 129 + k]);
    float s = 0.0f;
    for (int k = 0; k < 128; ++k) {
      float e = __expf(Lg[t * 129 + k] - mx);
      Lg[t * 129 + k] = e;
      s += e;
    }
    float inv = 1.0f / s;
    for (int k = 0; k < 128; ++k) Ps[t * 136 + k] = f2bf(Lg[t * 129 + k] * inv);
  }
  __syncthreads();
  // ---- out = P @ V ----
  if (wave < 2) {
    const int tq = wave;
    v8f acc = ZERO8;
#pragma unroll
    for (int kb = 0; kb < 4; ++kb) {
      v16bf ap = load_frag_l(Ps, 136, tq * 16, kb * 32);
      v16bf bv = load_frag_l(Vt, 136, 0, kb * 32);
      acc = WMMA_BF16(ap, bv, acc);
    }
    int d = lane & 15;
    gu16_p outg = (gu16_p)(void*)Out;
#pragma unroll
    for (int r = 0; r < 8; ++r) {
      int ql = tq * 16 + r + ((lane >> 4) << 3);
      int n = qh * 32 + ql;
      outg[((size_t)b * 128 + n) * 128 + h * 16 + d] = f2bf(acc[r]);
    }
  }
}

// ---------- kernel 3: output projection (bf16 A @ bf16 Wt + bias -> f32) ----------
__global__ __launch_bounds__(256) void gemm_out_kernel(
    const unsigned short* __restrict__ A, const unsigned short* __restrict__ Wt,
    const float* __restrict__ bias, float* __restrict__ Out) {
  __shared__ unsigned short As[128 * 136];
  const int t = threadIdx.x;
  const int row0 = blockIdx.x * 128;
  const u32x4* a4 = (const u32x4*)(A + (size_t)row0 * 128);
  for (int i = t; i < 2048; i += 256) {
    int r = i >> 4, kp = (i & 15) << 3;
    async_copy_b128(&As[r * 136 + kp], &a4[i]);
  }
  async_wait();
  __syncthreads();
  const int wave = t >> 5, lane = t & 31;
  const int tq = wave;
  v16bf af[4];
#pragma unroll
  for (int kb = 0; kb < 4; ++kb) af[kb] = load_frag_l(As, 136, tq * 16, kb * 32);
  gf_cp biasg = (gf_cp)(const void*)bias;
  gf_p outg = (gf_p)(void*)Out;
  for (int tc = 0; tc < 8; ++tc) {
    v8f acc = ZERO8;
#pragma unroll
    for (int kb = 0; kb < 4; ++kb) {
      v16bf bfrag = load_frag_g(Wt, 128, tc * 16, kb * 32);
      acc = WMMA_BF16(af[kb], bfrag, acc);
    }
    int col = tc * 16 + (lane & 15);
    float bv = biasg[col];
#pragma unroll
    for (int r = 0; r < 8; ++r) {
      int row = row0 + tq * 16 + r + ((lane >> 4) << 3);
      outg[(size_t)row * 128 + col] = acc[r] + bv;
    }
  }
}

// ---------- kernel 4: gate GEMM + sigmoid + blend ----------
__global__ __launch_bounds__(256) void gate_kernel(
    const float* __restrict__ Xs, const float* __restrict__ Os,
    const float* __restrict__ Oc, const unsigned short* __restrict__ Wtg,
    const float* __restrict__ bg, float* __restrict__ Y) {
  __shared__ unsigned short As[64 * 392];  // 50 KB, 64 rows x 384 (+pad)
  const int t = threadIdx.x;
  const int row0 = blockIdx.x * 64;
  gf4_cp xs4 = (gf4_cp)(const void*)Xs;
  gf4_cp os4 = (gf4_cp)(const void*)Os;
  gf4_cp oc4 = (gf4_cp)(const void*)Oc;
  for (int i = t; i < 64 * 32; i += 256) {
    int r = i >> 5, c = (i & 31) << 2;
    size_t src = ((size_t)(row0 + r) * 128 + c) >> 2;
    f32x4 vx = xs4[src];
    f32x4 vs = os4[src];
    f32x4 vc = oc4[src];
    unsigned short* p = &As[r * 392];
    p[c + 0] = f2bf(vx.x); p[c + 1] = f2bf(vx.y);
    p[c + 2] = f2bf(vx.z); p[c + 3] = f2bf(vx.w);
    p[128 + c + 0] = f2bf(vs.x); p[128 + c + 1] = f2bf(vs.y);
    p[128 + c + 2] = f2bf(vs.z); p[128 + c + 3] = f2bf(vs.w);
    p[256 + c + 0] = f2bf(vc.x); p[256 + c + 1] = f2bf(vc.y);
    p[256 + c + 2] = f2bf(vc.z); p[256 + c + 3] = f2bf(vc.w);
  }
  __syncthreads();
  const int wave = t >> 5, lane = t & 31;
  const int tq = wave >> 1;        // 0..3
  const int tc0 = (wave & 1) * 4;  // 0 or 4
  gf_cp bgg = (gf_cp)(const void*)bg;
  gf_cp osg = (gf_cp)(const void*)Os;
  gf_cp ocg = (gf_cp)(const void*)Oc;
  gf_p yg = (gf_p)(void*)Y;
  for (int tc = tc0; tc < tc0 + 4; ++tc) {
    v8f acc = ZERO8;
#pragma unroll
    for (int kb = 0; kb < 12; ++kb) {
      v16bf a = load_frag_l(As, 392, tq * 16, kb * 32);
      v16bf bfrag = load_frag_g(Wtg, 384, tc * 16, kb * 32);
      acc = WMMA_BF16(a, bfrag, acc);
    }
    int col = tc * 16 + (lane & 15);
    float bb = bgg[col];
#pragma unroll
    for (int r = 0; r < 8; ++r) {
      int row = row0 + tq * 16 + r + ((lane >> 4) << 3);
      float g = 1.0f / (1.0f + __expf(-(acc[r] + bb)));
      float os = osg[(size_t)row * 128 + col];
      float oc = ocg[(size_t)row * 128 + col];
      yg[(size_t)row * 128 + col] = (1.0f - g) * os + g * oc;
    }
  }
}

// ---------- host launcher ----------
extern "C" void kernel_launch(void* const* d_in, const int* in_sizes, int n_in,
                              void* d_out, int out_size, void* d_ws,
                              size_t ws_size, hipStream_t stream) {
  (void)in_sizes; (void)n_in; (void)out_size; (void)ws_size;
  const float* x_self = (const float*)d_in[0];
  const int* adj = (const int*)d_in[1];
  const int* mask_s = (const int*)d_in[2];
  const float* x_cross = (const float*)d_in[3];
  const int* mask_c = (const int*)d_in[4];
  // weights/biases: wq_s,bq_s,wk_s,bk_s,wv_s,bv_s,wo_s,bo_s,wq_c,...,wo_c,bo_c
  const float* W[8];
  const float* Bv[8];
  for (int i = 0; i < 8; ++i) {
    W[i]  = (const float*)d_in[5 + 2 * i];
    Bv[i] = (const float*)d_in[6 + 2 * i];
  }
  const float* wg = (const float*)d_in[21];
  const float* bg = (const float*)d_in[22];

  // workspace layout (bytes)
  char* ws = (char*)d_ws;
  unsigned short* wt[8];
  for (int i = 0; i < 8; ++i) wt[i] = (unsigned short*)(ws + (size_t)i * 32768);
  unsigned short* wtg = (unsigned short*)(ws + 262144);        // 384x128 bf16
  const size_t HEAD_SZ = 8388608;                              // [B,H,128,16] bf16
  unsigned short* q_s = (unsigned short*)(ws + 360448);
  unsigned short* k_s = (unsigned short*)(ws + 360448 + 1 * HEAD_SZ);
  unsigned short* v_s = (unsigned short*)(ws + 360448 + 2 * HEAD_SZ);
  unsigned short* q_c = (unsigned short*)(ws + 360448 + 3 * HEAD_SZ);
  unsigned short* k_c = (unsigned short*)(ws + 360448 + 4 * HEAD_SZ);
  unsigned short* v_c = (unsigned short*)(ws + 360448 + 5 * HEAD_SZ);
  unsigned short* attn_s = (unsigned short*)(ws + 50692096);   // [32768,128] bf16
  unsigned short* attn_c = (unsigned short*)(ws + 59080704);
  float* out_s = (float*)(ws + 67469312);                      // [32768,128] f32
  float* out_c = (float*)(ws + 84246528);

  // 0) convert weights to bf16 col-major
  for (int i = 0; i < 8; ++i)
    wconv_kernel<<<64, 256, 0, stream>>>(W[i], wt[i], 128, 128);
  wconv_kernel<<<192, 256, 0, stream>>>(wg, wtg, 384, 128);

  // 1) projections (q_c reads x_self)
  proj_qkv_kernel<<<256, 256, 0, stream>>>(
      x_self, wt[0], Bv[0], q_s, wt[1], Bv[1], k_s, wt[2], Bv[2], v_s,
      wt[4], Bv[4], q_c, 4);
  proj_qkv_kernel<<<256, 256, 0, stream>>>(
      x_cross, wt[5], Bv[5], k_c, wt[6], Bv[6], v_c,
      nullptr, nullptr, nullptr, nullptr, nullptr, nullptr, 2);

  // 2) attention (self uses adjacency, cross does not)
  attn_kernel<<<8192, 256, 0, stream>>>(q_s, k_s, v_s, adj, mask_s, attn_s, 1);
  attn_kernel<<<8192, 256, 0, stream>>>(q_c, k_c, v_c, nullptr, mask_c, attn_c, 0);

  // 3) output projections
  gemm_out_kernel<<<256, 256, 0, stream>>>(attn_s, wt[3], Bv[3], out_s);
  gemm_out_kernel<<<256, 256, 0, stream>>>(attn_c, wt[7], Bv[7], out_c);

  // 4) gate + blend
  gate_kernel<<<512, 256, 0, stream>>>(x_self, out_s, out_c, wtg, bg,
                                       (float*)d_out);
}
